// GDN_60155311947937
// MI455X (gfx1250) — matrix operand order repaired and checked by reading, not compiled, not run on hardware
//
#include <hip/hip_runtime.h>

// ---------------------------------------------------------------------------
// GDN forward for MI455X (gfx1250, wave32). GEMMs via v_wmma_f32_16x16x32_f16.
// Shapes: B=32, N=1000, Fin=DIM=64, TOPK=20. E = B*N*TOPK = 640k edges.
// Both GEMMs (embedding Gram matrix, x @ lin_w) consume B pre-transposed so
// every fragment load is a contiguous per-lane b128; K=64 fully unrolled.
// ---------------------------------------------------------------------------

#define BATCH 32
#define NN    1000          // nodes
#define NP    1008          // nodes padded to multiple of 16 for WMMA tiles
#define DIM   64
#define ROWS  (BATCH * NN)  // 32000 = 125 * 256
#define TOPK  20
#define NEG_SLOPE 0.2f
#define BN_EPS 1e-5f

typedef __attribute__((ext_vector_type(16))) _Float16 v16h;
typedef __attribute__((ext_vector_type(8)))  _Float16 v8h;
typedef __attribute__((ext_vector_type(8)))  float    v8f;

// ---------------------------------------------------------------------------
// 0) zero small stats buffer
// ---------------------------------------------------------------------------
__global__ void zero_kernel(float* p, int n) {
    int i = blockIdx.x * 256 + threadIdx.x;
    if (i < n) p[i] = 0.0f;
}

// ---------------------------------------------------------------------------
// 1) prep: f32->f16 conversions + row-normalized (cosine) embedding, padded.
//    idx < ROWS*DIM : data  -> xf16
//    idx < DIM*DIM  : lin_w -> wt16 TRANSPOSED (wt16[n][k] = lin_w[k][n])
//    idx < NP       : emb row -> ehat (L2-normalized f16; pad rows zero)
// ---------------------------------------------------------------------------
__global__ __launch_bounds__(256)
void prep_kernel(const float* __restrict__ data, const float* __restrict__ emb,
                 const float* __restrict__ lin_w,
                 _Float16* __restrict__ ehat, _Float16* __restrict__ wt16,
                 _Float16* __restrict__ xf16) {
    int idx = blockIdx.x * 256 + threadIdx.x;
    if (idx < ROWS * DIM) xf16[idx] = (_Float16)data[idx];
    if (idx < DIM * DIM) {
        const int n = idx >> 6, k = idx & 63;        // wt16 is N x K row-major
        wt16[idx] = (_Float16)lin_w[k * DIM + n];
    }
    if (idx < NP) {
        _Float16* er = ehat + idx * DIM;
        if (idx < NN) {
            const float* e = emb + idx * DIM;
            float s = 0.f;
            #pragma unroll 8
            for (int d = 0; d < DIM; ++d) s += e[d] * e[d];
            float inv = rsqrtf(s);
            for (int d = 0; d < DIM; ++d) er[d] = (_Float16)(e[d] * inv);
        } else {
            for (int d = 0; d < DIM; ++d) er[d] = (_Float16)0.f;
        }
    }
}

// ---------------------------------------------------------------------------
// 2) One-wave WMMA GEMM, K = 64 (two unrolled 16x16x32 steps):
//      C[16x16 tile] = A(M x 64, row-major) * Bt^T  (Bt is N x 64 row-major)
//    ISA fragment layouts (CDNA5 ISA):
//      A 16x32 f16 : lane half selects K {0..7,16..23} / {8..15,24..31},
//                    row = lane & 15  -> two contiguous 16B chunks per lane
//      B 32x16 f16 : lane half selects K {0..7,16..23} / {8..15,24..31}, col = lane & 15
//                    -> with Bt row-major: two contiguous 16B chunks per lane
//      C 16x16 f32 : VGPR v -> M = v + 8*(lane>>4), N = lane & 15
// ---------------------------------------------------------------------------
__global__ __launch_bounds__(32)
void wmma_gemm_kernel(const _Float16* __restrict__ A, int lda,
                      const _Float16* __restrict__ Bt, int ldb,
                      float* __restrict__ C, int ldc) {
    const int lane = threadIdx.x;
    const int row  = lane & 15;
    const int hi   = lane >> 4;                     // 0 or 1 (lane half)
    const int mbase = blockIdx.x * 16;
    const int nbase = blockIdx.y * 16;

    const _Float16* Ap = A  + (mbase + row) * lda + hi * 8;    // koffA = hi*8
    const _Float16* Bp = Bt + (nbase + row) * ldb + hi * 8;    // koffB = hi*8

    v8f acc = {};
    #pragma unroll
    for (int kk = 0; kk < 64; kk += 32) {
        __builtin_prefetch(Ap + kk + 32, 0, 1);     // -> global_prefetch_b8
        v8h a0 = *(const v8h*)(Ap + kk);            // K chunk {0..7}+koffA
        v8h a1 = *(const v8h*)(Ap + kk + 16);       // K chunk {16..23}+koffA
        v16h a, b;
        #pragma unroll
        for (int v = 0; v < 8; ++v) { a[v] = a0[v]; a[v + 8] = a1[v]; }
        v8h b0 = *(const v8h*)(Bp + kk);            // K chunk {0..7}+koffB
        v8h b1 = *(const v8h*)(Bp + kk + 16);       // K chunk {16..23}+koffB
        #pragma unroll
        for (int v = 0; v < 8; ++v) { b[v] = b0[v]; b[v + 8] = b1[v]; }
        acc = __builtin_amdgcn_wmma_f32_16x16x32_f16(
            /*neg_a=*/false, a, /*neg_b=*/false, b,
            /*c_mod=*/(short)0, acc, /*reuse_a=*/false, /*reuse_b=*/false);
    }
    #pragma unroll
    for (int v = 0; v < 8; ++v) {
        const int r = mbase + v + 8 * hi;
        C[r * ldc + nbase + row] = acc[v];
    }
}

// ---------------------------------------------------------------------------
// 3) top-20 per cosine row: 20 rounds of block-wide argmax + mask (-inf).
// ---------------------------------------------------------------------------
__global__ __launch_bounds__(256)
void topk_kernel(float* __restrict__ cosm, int ldc, int* __restrict__ topk) {
    const int i = blockIdx.x, tid = threadIdx.x;
    __shared__ float sv[256];
    __shared__ int   si[256];
    float* row = cosm + (long)i * ldc;
    for (int t = 0; t < TOPK; ++t) {
        float bv = -1e30f; int bi = 0x7fffffff;
        for (int j = tid; j < NN; j += 256) {
            float v = row[j];
            if (v > bv || (v == bv && j < bi)) { bv = v; bi = j; }
        }
        sv[tid] = bv; si[tid] = bi;
        __syncthreads();
        for (int s = 128; s > 0; s >>= 1) {
            if (tid < s) {
                if (sv[tid + s] > sv[tid] ||
                    (sv[tid + s] == sv[tid] && si[tid + s] < si[tid])) {
                    sv[tid] = sv[tid + s]; si[tid] = si[tid + s];
                }
            }
            __syncthreads();
        }
        if (tid == 0) { topk[i * TOPK + t] = si[0]; row[si[0]] = -1e30f; }
        __syncthreads();
    }
}

// ---------------------------------------------------------------------------
// 4) per-node attention scalars: alpha(e) = ai[tgt] + aj[src]
// ---------------------------------------------------------------------------
__global__ __launch_bounds__(256)
void node_scalar_kernel(const float* __restrict__ xl, const float* __restrict__ emb,
                        const float* __restrict__ att_i, const float* __restrict__ att_j,
                        const float* __restrict__ att_em_i, const float* __restrict__ att_em_j,
                        float* __restrict__ ai, float* __restrict__ aj) {
    const int bn = blockIdx.x * 256 + threadIdx.x;
    if (bn >= ROWS) return;
    const int n = bn % NN;
    const float* xr = xl + bn * DIM;
    const float* er = emb + n * DIM;
    float si = 0.f, sj = 0.f;
    #pragma unroll 8
    for (int d = 0; d < DIM; ++d) {
        si += xr[d] * att_i[d] + er[d] * att_em_i[d];
        sj += xr[d] * att_j[d] + er[d] * att_em_j[d];
    }
    ai[bn] = si; aj[bn] = sj;
}

// ---------------------------------------------------------------------------
// 5) per-node edge softmax + aggregation (one 64-thread block per node)
// ---------------------------------------------------------------------------
__global__ __launch_bounds__(64)
void aggregate_kernel(const float* __restrict__ xl, const float* __restrict__ ai,
                      const float* __restrict__ aj, const int* __restrict__ topk,
                      const float* __restrict__ bias, float* __restrict__ agg) {
    const int bn = blockIdx.x;
    const int b = bn / NN, n = bn - b * NN;
    const int tid = threadIdx.x;
    __shared__ float sa[TOPK];
    __shared__ int   ssrc[TOPK];
    __shared__ float sden;
    if (tid < TOPK) {
        const int sb = b * NN + topk[n * TOPK + tid];
        ssrc[tid] = sb;
        float al = ai[bn] + aj[sb];
        sa[tid] = al > 0.f ? al : NEG_SLOPE * al;    // leaky_relu
    }
    __syncthreads();
    if (tid == 0) {                                  // segment softmax (20 edges)
        float m = sa[0];
        for (int e = 1; e < TOPK; ++e) m = fmaxf(m, sa[e]);
        float den = 0.f;
        for (int e = 0; e < TOPK; ++e) { float t = expf(sa[e] - m); sa[e] = t; den += t; }
        sden = den + 1e-16f;
    }
    __syncthreads();
    const int d = tid;
    float acc = 0.f;
    #pragma unroll 4
    for (int e = 0; e < TOPK; ++e) acc += sa[e] * xl[ssrc[e] * DIM + d];
    agg[bn * DIM + d] = acc / sden + bias[d];
}

// ---------------------------------------------------------------------------
// 6) column stats (sum / sumsq) over 32000 rows; mode 1 recomputes the
//    BN1+ReLU+emb-mul value on the fly (avoids materializing h).
// ---------------------------------------------------------------------------
__global__ __launch_bounds__(256)
void stats_kernel(const float* __restrict__ src, const float* __restrict__ emb,
                  const float* __restrict__ mu1, const float* __restrict__ var1,
                  const float* __restrict__ g1, const float* __restrict__ b1,
                  float* __restrict__ sum, float* __restrict__ sq, int mode) {
    const int d = threadIdx.x & 63, s = threadIdx.x >> 6;   // 64 cols x 4 slices
    const int base = blockIdx.x * 256;
    float ls = 0.f, lq = 0.f;
    for (int r0 = s; r0 < 256; r0 += 4) {
        const int r = base + r0;
        float v = src[r * DIM + d];
        if (mode) {
            v = (v - mu1[d]) * rsqrtf(var1[d] + BN_EPS) * g1[d] + b1[d];
            v = fmaxf(v, 0.f);
            v *= emb[(r % NN) * DIM + d];
        }
        ls += v; lq += v * v;
    }
    __shared__ float ss[4][64], sqq[4][64];
    ss[s][d] = ls; sqq[s][d] = lq;
    __syncthreads();
    if (s == 0) {
        atomicAdd(&sum[d], ss[0][d] + ss[1][d] + ss[2][d] + ss[3][d]);
        atomicAdd(&sq[d],  sqq[0][d] + sqq[1][d] + sqq[2][d] + sqq[3][d]);
    }
}

__global__ __launch_bounds__(64)
void finalize_stats_kernel(const float* __restrict__ sum, const float* __restrict__ sq,
                           float* __restrict__ mu, float* __restrict__ var) {
    const int d = threadIdx.x;
    const float inv = 1.0f / (float)ROWS;
    const float m = sum[d] * inv;
    mu[d] = m;
    var[d] = sq[d] * inv - m * m;
}

// ---------------------------------------------------------------------------
// 7) fused BN1 -> ReLU -> *emb -> BN2 -> ReLU -> head dot ( [64] -> [1] )
// ---------------------------------------------------------------------------
__global__ __launch_bounds__(256)
void final_kernel(const float* __restrict__ agg, const float* __restrict__ emb,
                  const float* __restrict__ mu1, const float* __restrict__ var1,
                  const float* __restrict__ g1, const float* __restrict__ b1,
                  const float* __restrict__ mu2, const float* __restrict__ var2,
                  const float* __restrict__ g2, const float* __restrict__ b2,
                  const float* __restrict__ out_w, const float* __restrict__ out_b,
                  float* __restrict__ out) {
    const int bn = blockIdx.x * 256 + threadIdx.x;
    if (bn >= ROWS) return;
    const int n = bn % NN;
    float acc = out_b[0];
    #pragma unroll 8
    for (int d = 0; d < DIM; ++d) {
        float v = agg[bn * DIM + d];
        v = (v - mu1[d]) * rsqrtf(var1[d] + BN_EPS) * g1[d] + b1[d];
        v = fmaxf(v, 0.f);
        v *= emb[n * DIM + d];
        v = (v - mu2[d]) * rsqrtf(var2[d] + BN_EPS) * g2[d] + b2[d];
        v = fmaxf(v, 0.f);
        acc += v * out_w[d];
    }
    out[bn] = acc;
}

// ---------------------------------------------------------------------------
// host launcher
// ---------------------------------------------------------------------------
extern "C" void kernel_launch(void* const* d_in, const int* in_sizes, int n_in,
                              void* d_out, int out_size, void* d_ws, size_t ws_size,
                              hipStream_t stream) {
    const float* data     = (const float*)d_in[0];
    const float* emb      = (const float*)d_in[1];
    const float* lin_w    = (const float*)d_in[2];
    const float* att_i    = (const float*)d_in[3];
    const float* att_j    = (const float*)d_in[4];
    const float* att_em_i = (const float*)d_in[5];
    const float* att_em_j = (const float*)d_in[6];
    const float* gnn_bias = (const float*)d_in[7];
    const float* g1       = (const float*)d_in[8];
    const float* b1       = (const float*)d_in[9];
    const float* g2       = (const float*)d_in[10];
    const float* b2       = (const float*)d_in[11];
    const float* out_w    = (const float*)d_in[12];
    const float* out_b    = (const float*)d_in[13];
    float* out = (float*)d_out;

    // ---- workspace carve-up (~25 MB total) ----
    char* ws = (char*)d_ws;
    size_t off = 0;
    auto take = [&](size_t bytes) {
        size_t r = off;
        off += (bytes + 255) & ~(size_t)255;
        return r;
    };
    _Float16* ehat = (_Float16*)(ws + take((size_t)NP * DIM * 2));       // 129 KB
    _Float16* wt16 = (_Float16*)(ws + take((size_t)DIM * DIM * 2));      // 8 KB
    _Float16* xf16 = (_Float16*)(ws + take((size_t)ROWS * DIM * 2));     // 4.1 MB
    float* cosm    = (float*)(ws + take((size_t)NP * NP * 4));           // 4.1 MB
    int*   topk    = (int*)  (ws + take((size_t)NN * TOPK * 4));         // 80 KB
    float* xl      = (float*)(ws + take((size_t)ROWS * DIM * 4));        // 8.2 MB
    float* ai      = (float*)(ws + take((size_t)ROWS * 4));
    float* aj      = (float*)(ws + take((size_t)ROWS * 4));
    float* agg     = (float*)(ws + take((size_t)ROWS * DIM * 4));        // 8.2 MB
    float* stats   = (float*)(ws + take((size_t)8 * DIM * 4));           // 2 KB
    float* sum1 = stats + 0 * DIM, *sq1 = stats + 1 * DIM;
    float* sum2 = stats + 2 * DIM, *sq2 = stats + 3 * DIM;
    float* mu1  = stats + 4 * DIM, *var1 = stats + 5 * DIM;
    float* mu2  = stats + 6 * DIM, *var2 = stats + 7 * DIM;
    (void)in_sizes; (void)n_in; (void)out_size; (void)ws_size;

    // 0) zero stats accumulators
    zero_kernel<<<1, 256, 0, stream>>>(stats, 4 * DIM);
    // 1) conversions + transposed lin_w + normalized embedding
    prep_kernel<<<(ROWS * DIM) / 256, 256, 0, stream>>>(data, emb, lin_w, ehat, wt16, xf16);
    // 2) cosine Gram matrix via WMMA: cos = Ehat * Ehat^T   (1008x1008, K=64)
    wmma_gemm_kernel<<<dim3(NP / 16, NP / 16), 32, 0, stream>>>(
        ehat, DIM, ehat, DIM, cosm, NP);
    // 3) top-20 neighbors per node
    topk_kernel<<<NN, 256, 0, stream>>>(cosm, NP, topk);
    // 4) xl = data @ lin_w via WMMA  (32000x64, K=64, B pre-transposed)
    wmma_gemm_kernel<<<dim3(ROWS / 16, DIM / 16), 32, 0, stream>>>(
        xf16, DIM, wt16, DIM, xl, DIM);
    // 5) per-node attention scalars
    node_scalar_kernel<<<ROWS / 256, 256, 0, stream>>>(
        xl, emb, att_i, att_j, att_em_i, att_em_j, ai, aj);
    // 6) edge softmax + aggregation + bias
    aggregate_kernel<<<ROWS, 64, 0, stream>>>(xl, ai, aj, topk, gnn_bias, agg);
    // 7) BN1 stats
    stats_kernel<<<ROWS / 256, 256, 0, stream>>>(agg, emb, mu1, var1, g1, b1,
                                                 sum1, sq1, /*mode=*/0);
    finalize_stats_kernel<<<1, 64, 0, stream>>>(sum1, sq1, mu1, var1);
    // 8) BN2 stats on h = relu(bn1(agg)) * emb  (recomputed on the fly)
    stats_kernel<<<ROWS / 256, 256, 0, stream>>>(agg, emb, mu1, var1, g1, b1,
                                                 sum2, sq2, /*mode=*/1);
    finalize_stats_kernel<<<1, 64, 0, stream>>>(sum2, sq2, mu2, var2);
    // 9) fused BN1/ReLU/mul/BN2/ReLU/head
    final_kernel<<<ROWS / 256, 256, 0, stream>>>(agg, emb, mu1, var1, g1, b1,
                                                 mu2, var2, g2, b2, out_w, out_b, out);
}
